// Head_69166153334823
// MI455X (gfx1250) — matrix operand order repaired and checked by reading, not compiled, not run on hardware
//
#include <hip/hip_runtime.h>

#define BATCH 4
#define TSEQ  4096
#define CEMB  384
#define HDIM  64

typedef __attribute__((ext_vector_type(16))) __bf16 v16bf;
typedef __attribute__((ext_vector_type(8)))  float  v8f;

struct U4x2 { uint4 lo, hi; };
struct Frag8 { v16bf f[8]; };   // 8 B-fragments (one 64-key block of K or V)

// round-to-nearest-even f32 -> bf16 via integer path
__device__ __forceinline__ __bf16 f2bf(float f) {
  unsigned u = __builtin_bit_cast(unsigned, f);
  u += 0x7FFFu + ((u >> 16) & 1u);
  unsigned short h = (unsigned short)(u >> 16);
  return __builtin_bit_cast(__bf16, h);
}

// A-matrix fragment (16x32 bf16): lane holds row (lane&15); 16 contiguous K elems at koff
__device__ __forceinline__ v16bf load_afrag(const __bf16* row, int koff) {
  U4x2 t;
  t.lo = *(const uint4*)(row + koff);
  t.hi = *(const uint4*)(row + koff + 8);
  return __builtin_bit_cast(v16bf, t);
}

// B-matrix fragment (32x16 bf16): lane holds col (lane&15); 16 contiguous K elems
__device__ __forceinline__ v16bf load_bfrag(const __bf16* p) {
  U4x2 t;
  t.lo = *(const uint4*)(p);
  t.hi = *(const uint4*)(p + 8);
  return __builtin_bit_cast(v16bf, t);
}

__device__ __forceinline__ v8f wmma_bf16(v16bf a, v16bf b, v8f c) {
  return __builtin_amdgcn_wmma_f32_16x16x32_bf16(false, a, false, b,
                                                 (short)0, c, false, false);
}

// Load one 64-key block of K as 8 B-fragments (rows contiguous in head dim)
__device__ __forceinline__ Frag8 loadK(const __bf16* Kbt, int kbase, int col, int bkoff) {
  Frag8 k;
#pragma unroll
  for (int j = 0; j < 4; ++j) {
    const __bf16* krow = Kbt + (size_t)(kbase + j * 16 + col) * HDIM;
    k.f[2 * j]     = load_bfrag(krow + bkoff);
    k.f[2 * j + 1] = load_bfrag(krow + 32 + bkoff);
  }
  return k;
}

// Load one 64-key block of V^T as 8 B-fragments (key dim contiguous)
__device__ __forceinline__ Frag8 loadV(const __bf16* Vbt, int kbase, int col, int bkoff) {
  Frag8 v;
#pragma unroll
  for (int nt = 0; nt < 4; ++nt) {
    const __bf16* vrow = Vbt + (size_t)(nt * 16 + col) * TSEQ + kbase;
    v.f[2 * nt]     = load_bfrag(vrow + bkoff);
    v.f[2 * nt + 1] = load_bfrag(vrow + 32 + bkoff);
  }
  return v;
}

// ---------------------------------------------------------------------------
// Kernel 1a: streaming convert x [B*T,384] f32 -> bf16.
// ---------------------------------------------------------------------------
__global__ void head_prep_x(const float* __restrict__ x,
                            __bf16* __restrict__ xb, int total) {
  int idx = blockIdx.x * blockDim.x + threadIdx.x;
  if (idx < total) xb[idx] = f2bf(x[idx]);
}

// ---------------------------------------------------------------------------
// Kernel 1b: transpose + quantize weights [384,64]f32 -> Wt[3][64][384] bf16.
// ---------------------------------------------------------------------------
__global__ void head_prep_weights(const float* __restrict__ Wk,
                                  const float* __restrict__ Wq,
                                  const float* __restrict__ Wv,
                                  __bf16* __restrict__ Wt) {
  int idx = blockIdx.x * blockDim.x + threadIdx.x;
  const int total = 3 * HDIM * CEMB;
  if (idx >= total) return;
  int w   = idx / (HDIM * CEMB);
  int rem = idx % (HDIM * CEMB);
  int n   = rem / CEMB;
  int c   = rem % CEMB;
  const float* W = (w == 0) ? Wk : (w == 1) ? Wq : Wv;
  Wt[idx] = f2bf(W[c * HDIM + n]);
}

// ---------------------------------------------------------------------------
// Kernel 2: projections. One wave per 16-row tile of xb. K,Q stored [B*T,64]
// bf16 (Q pre-scaled by C^-0.5), V stored transposed [B][64][T] bf16.
// ---------------------------------------------------------------------------
__global__ __launch_bounds__(32) void head_project(
    const __bf16* __restrict__ xb,   // [B*T, 384] bf16
    const __bf16* __restrict__ Wt,   // [3][64][384]
    __bf16* __restrict__ Kb,         // [B*T, 64]
    __bf16* __restrict__ Qb,         // [B*T, 64]
    __bf16* __restrict__ Vt)         // [B][64][T]
{
  const int lane    = threadIdx.x;
  const int rowbase = blockIdx.x * 16;
  const int col     = lane & 15;
  const int hi      = lane >> 4;
  const int akoff   = hi * 16;
  const int bkoff   = hi * 16;
  const __bf16* xrow = xb + (size_t)(rowbase + col) * CEMB;

  v8f zero = {};
  v8f acc[3][4];
#pragma unroll
  for (int w = 0; w < 3; ++w)
#pragma unroll
    for (int nt = 0; nt < 4; ++nt) acc[w][nt] = zero;

  for (int kb = 0; kb < CEMB; kb += 32) {
    v16bf a = load_afrag(xrow + kb, akoff);
#pragma unroll
    for (int w = 0; w < 3; ++w) {
#pragma unroll
      for (int nt = 0; nt < 4; ++nt) {
        const __bf16* brow = Wt + (size_t)(w * HDIM + nt * 16 + col) * CEMB + kb + bkoff;
        acc[w][nt] = wmma_bf16(a, load_bfrag(brow), acc[w][nt]);
      }
    }
  }

  const float qscale = 0.05103103630798287f;  // 384^-0.5
  const int b  = rowbase / TSEQ;
  const int t0 = rowbase % TSEQ;
#pragma unroll
  for (int nt = 0; nt < 4; ++nt) {
#pragma unroll
    for (int r = 0; r < 8; ++r) {
      const int row = rowbase + r + 8 * hi;
      const int n   = nt * 16 + col;
      const int t   = t0 + r + 8 * hi;
      Kb[(size_t)row * HDIM + n] = f2bf(acc[0][nt][r]);
      Qb[(size_t)row * HDIM + n] = f2bf(acc[1][nt][r] * qscale);
      Vt[((size_t)b * HDIM + n) * TSEQ + t] = f2bf(acc[2][nt][r]);
    }
  }
}

// ---------------------------------------------------------------------------
// Kernel 3: causal flash attention, software-pipelined. One wave handles 16
// query rows; key blocks of 64. K fragments double-buffered in registers and
// V fragments issued before the softmax so the ~150-instruction VALU stretch
// hides the L2 round trip (occupancy is 1 wave/SIMD -> ILP is the only
// latency hiding available).
// ---------------------------------------------------------------------------
__global__ __launch_bounds__(32) void head_flash(
    const __bf16* __restrict__ Qb,   // [B*T,64], pre-scaled
    const __bf16* __restrict__ Kb,   // [B*T,64]
    const __bf16* __restrict__ Vt,   // [B][64][T]
    float* __restrict__ out)         // [B*T,64]
{
  __shared__ __bf16 lds_p[16][64];   // 2 KB P tile

  const int lane  = threadIdx.x;
  const int nqb   = TSEQ / 16;
  const int b     = blockIdx.x / nqb;
  const int qbase = (blockIdx.x % nqb) * 16;
  const int col   = lane & 15;
  const int hi    = lane >> 4;
  const int akoff = hi * 16;
  const int bkoff = hi * 16;

  const __bf16* Qbt = Qb + (size_t)b * TSEQ * HDIM;
  const __bf16* Kbt = Kb + (size_t)b * TSEQ * HDIM;
  const __bf16* Vbt = Vt + (size_t)b * HDIM * TSEQ;

  // Q fragments for head-dim halves 0..31 and 32..63 (reused all iterations)
  const __bf16* qrowp = Qbt + (size_t)(qbase + col) * HDIM;
  v16bf qa0 = load_afrag(qrowp, akoff);
  v16bf qa1 = load_afrag(qrowp + 32, akoff);

  const float NEG_INF = -__builtin_inff();
  float m[8], l[8];
  v8f zero = {};
  v8f o[4];
#pragma unroll
  for (int r = 0; r < 8; ++r) { m[r] = NEG_INF; l[r] = 0.0f; }
#pragma unroll
  for (int nt = 0; nt < 4; ++nt) o[nt] = zero;

  const int kend = qbase + 16;          // exclusive causal key limit
  const int nkb  = (kend + 63) / 64;    // key blocks of 64

  Frag8 kcur = loadK(Kbt, 0, col, bkoff);   // prologue: block 0 K fragments

#pragma unroll 2
  for (int kbi = 0; kbi < nkb; ++kbi) {
    const int kbase = kbi * 64;

    // S = Q K^T : four 16-key column tiles, K-dim = 64 via two x32 WMMAs each
    v8f s[4];
#pragma unroll
    for (int j = 0; j < 4; ++j)
      s[j] = wmma_bf16(qa1, kcur.f[2 * j + 1], wmma_bf16(qa0, kcur.f[2 * j], zero));

    // issue this block's V loads and next block's K loads NOW; the softmax
    // VALU stretch below covers their latency.
    Frag8 vf = loadV(Vbt, kbase, col, bkoff);
    const int nkbase = (kbi + 1 < nkb) ? (kbase + 64) : kbase;  // clamp in-bounds
    Frag8 knext = loadK(Kbt, nkbase, col, bkoff);
    if (kbi + 1 < nkb) {
      __builtin_prefetch(Kbt + (size_t)(kbase + 128 + col) * HDIM, 0, 3);
      __builtin_prefetch(Vbt + (size_t)col * TSEQ + kbase + 64, 0, 3);
    }

    // causal mask + online softmax (rows 0-7 in lanes 0-15, 8-15 in 16-31)
#pragma unroll
    for (int r = 0; r < 8; ++r) {
      const int qrow = qbase + r + 8 * hi;
#pragma unroll
      for (int j = 0; j < 4; ++j)
        if (kbase + j * 16 + col > qrow) s[j][r] = NEG_INF;

      float v = fmaxf(fmaxf(s[0][r], s[1][r]), fmaxf(s[2][r], s[3][r]));
#pragma unroll
      for (int off = 1; off < 16; off <<= 1)
        v = fmaxf(v, __shfl_xor(v, off, 32));
      const float mnew  = fmaxf(m[r], v);
      const float alpha = __expf(m[r] - mnew);
      m[r] = mnew;

      float ps = 0.0f;
#pragma unroll
      for (int j = 0; j < 4; ++j) {
        const float p = __expf(s[j][r] - mnew);
        s[j][r] = p;
        ps += p;
      }
#pragma unroll
      for (int off = 1; off < 16; off <<= 1)
        ps += __shfl_xor(ps, off, 32);
      l[r] = l[r] * alpha + ps;

#pragma unroll
      for (int nt = 0; nt < 4; ++nt) o[nt][r] *= alpha;
    }

    // P: C-layout regs -> LDS (row-major 16x64) -> two A-layout fragments
#pragma unroll
    for (int r = 0; r < 8; ++r) {
#pragma unroll
      for (int j = 0; j < 4; ++j)
        lds_p[r + 8 * hi][j * 16 + col] = f2bf(s[j][r]);
    }
    asm volatile("s_wait_dscnt 0" ::: "memory");
    v16bf pa0 = load_afrag(&lds_p[col][0],  akoff);   // keys kbase+0..31
    v16bf pa1 = load_afrag(&lds_p[col][32], akoff);   // keys kbase+32..63
    asm volatile("" ::: "memory");

    // O += P V  (V transposed: key dim contiguous), K-dim = 64 via two x32
#pragma unroll
    for (int nt = 0; nt < 4; ++nt)
      o[nt] = wmma_bf16(pa1, vf.f[2 * nt + 1], wmma_bf16(pa0, vf.f[2 * nt], o[nt]));

    kcur = knext;   // rotate double buffer (renamed away by unroll-by-2)
  }

  // epilogue: normalize and store fp32
#pragma unroll
  for (int r = 0; r < 8; ++r) {
    const float inv  = 1.0f / l[r];
    const int   qrow = qbase + r + 8 * hi;
#pragma unroll
    for (int nt = 0; nt < 4; ++nt)
      out[((size_t)b * TSEQ + qrow) * HDIM + nt * 16 + col] = o[nt][r] * inv;
  }
}

// ---------------------------------------------------------------------------
extern "C" void kernel_launch(void* const* d_in, const int* in_sizes, int n_in,
                              void* d_out, int out_size, void* d_ws, size_t ws_size,
                              hipStream_t stream) {
  const float* x  = (const float*)d_in[0];
  const float* Wk = (const float*)d_in[1];
  const float* Wq = (const float*)d_in[2];
  const float* Wv = (const float*)d_in[3];
  float* out = (float*)d_out;

  size_t off = 0;
  auto carve = [&](size_t bytes) -> char* {
    char* p = (char*)d_ws + off;
    off += (bytes + 255) & ~(size_t)255;
    return p;
  };
  __bf16* Wt = (__bf16*)carve((size_t)3 * HDIM * CEMB * 2);          // 144 KB
  __bf16* xb = (__bf16*)carve((size_t)BATCH * TSEQ * CEMB * 2);      // 12.6 MB
  __bf16* Kb = (__bf16*)carve((size_t)BATCH * TSEQ * HDIM * 2);      // 2 MB
  __bf16* Qb = (__bf16*)carve((size_t)BATCH * TSEQ * HDIM * 2);      // 2 MB
  __bf16* Vt = (__bf16*)carve((size_t)BATCH * HDIM * TSEQ * 2);      // 2 MB
  (void)ws_size; (void)in_sizes; (void)n_in; (void)out_size;

  const int xtotal = BATCH * TSEQ * CEMB;
  head_prep_x<<<(xtotal + 255) / 256, 256, 0, stream>>>(x, xb, xtotal);

  const int wtotal = 3 * HDIM * CEMB;
  head_prep_weights<<<(wtotal + 255) / 256, 256, 0, stream>>>(Wk, Wq, Wv, Wt);

  const int ntiles = (BATCH * TSEQ) / 16;  // 1024 single-wave workgroups
  head_project<<<ntiles, 32, 0, stream>>>(xb, Wt, Kb, Qb, Vt);
  head_flash<<<ntiles, 32, 0, stream>>>(Qb, Kb, Vt, out);
}
